// LSTM_33878702030976
// MI455X (gfx1250) — compile-verified, matrix operand order checked by reading, not run on hardware
//
#include <hip/hip_runtime.h>
#include <hip/hip_fp16.h>

typedef _Float16 f16;
typedef __attribute__((ext_vector_type(16))) _Float16 v16h;
typedef __attribute__((ext_vector_type(8)))  _Float16 v8h;
typedef __attribute__((ext_vector_type(8)))  float    v8f;
typedef int i32x4 __attribute__((vector_size(16)));
typedef __attribute__((address_space(1))) i32x4* gp128;  // global int4*
typedef __attribute__((address_space(3))) i32x4* lp128;  // LDS int4*

#define B_    16384
#define T_    204
#define IN_   8
#define H_    3
#define LIN_  50
#define TH_   612   // T*H (= flatten width = out width)
#define KP_   640   // padded K for fc1  (20 chunks of 32)
#define LINP_ 64    // padded LIN        (2 chunks of 32)
#define NJP_  624   // padded T*NOUT rows of W_out (39 tiles of 16)

#define SWH_BYTES  (LINP_ * LINP_ * 2)          // 8192
#define SWO_BYTES  (NJP_ * LINP_ * 2)           // 79872
#define SY_BYTES   (16 * 64 * 2)                // 2048 per wave
#define SMEM_BYTES (SWH_BYTES + SWO_BYTES + 4 * SY_BYTES)  // 96256

#if defined(__HIP_DEVICE_COMPILE__) && \
    __has_builtin(__builtin_amdgcn_global_load_async_to_lds_b128) && \
    __has_builtin(__builtin_amdgcn_s_wait_asynccnt)
#define ASYNC_LDS 1
#else
#define ASYNC_LDS 0
#endif

// ---------------------------------------------------------------- prep: f32 -> f16 padded weights
__global__ void prep_weights(const float* __restrict__ Wfc1, const float* __restrict__ Whid,
                             const float* __restrict__ Wout,
                             f16* __restrict__ w1, f16* __restrict__ wh, f16* __restrict__ wo) {
  const int total = LINP_*KP_ + LINP_*LINP_ + NJP_*LINP_;
  for (int i = blockIdx.x * blockDim.x + threadIdx.x; i < total; i += gridDim.x * blockDim.x) {
    if (i < LINP_*KP_) {
      int r = i / KP_, k = i % KP_;
      w1[i] = (r < LIN_ && k < TH_) ? (f16)Wfc1[r*TH_ + k] : (f16)0.f;
    } else if (i < LINP_*KP_ + LINP_*LINP_) {
      int j = i - LINP_*KP_; int r = j / LINP_, k = j % LINP_;
      wh[j] = (r < LIN_ && k < LIN_) ? (f16)Whid[r*LIN_ + k] : (f16)0.f;
    } else {
      int j = i - LINP_*KP_ - LINP_*LINP_; int r = j / LINP_, k = j % LINP_;
      wo[j] = (r < TH_ && k < LIN_) ? (f16)Wout[r*LIN_ + k] : (f16)0.f;
    }
  }
}

// ---------------------------------------------------------------- fused 3-layer LSTM, 1 lane / sample
__device__ __forceinline__ float sigf(float v) { return 1.0f / (1.0f + __expf(-v)); }
__device__ __forceinline__ float tanh_fast(float v) {
  v = fminf(fmaxf(v, -15.f), 15.f);
  float e = __expf(2.0f * v);
  return (e - 1.0f) / (e + 1.0f);
}
__device__ __forceinline__ void lstm_cell(const float* g, float* h, float* c) {
#pragma unroll
  for (int j = 0; j < 3; ++j) {
    float ig = sigf(g[j]);
    float fg = sigf(g[3 + j]);
    float gg = tanh_fast(g[6 + j]);
    float og = sigf(g[9 + j]);
    float cn = fg * c[j] + ig * gg;
    c[j] = cn;
    h[j] = og * tanh_fast(cn);
  }
}

__global__ __launch_bounds__(64) void lstm3(const float* __restrict__ x,
    const float* __restrict__ Wih0, const float* __restrict__ Whh0,
    const float* __restrict__ bih0, const float* __restrict__ bhh0,
    const float* __restrict__ Wih1, const float* __restrict__ Whh1,
    const float* __restrict__ bih1, const float* __restrict__ bhh1,
    const float* __restrict__ Wih2, const float* __restrict__ Whh2,
    const float* __restrict__ bih2, const float* __restrict__ bhh2,
    f16* __restrict__ F) {
  __shared__ float sW[324];
  for (int i = threadIdx.x; i < 324; i += blockDim.x) {
    float v;
    if      (i <  96) v = Wih0[i];
    else if (i < 132) v = Whh0[i - 96];
    else if (i < 144) v = bih0[i - 132] + bhh0[i - 132];
    else if (i < 180) v = Wih1[i - 144];
    else if (i < 216) v = Whh1[i - 180];
    else if (i < 228) v = bih1[i - 216] + bhh1[i - 216];
    else if (i < 264) v = Wih2[i - 228];
    else if (i < 300) v = Whh2[i - 264];
    else              v = bih2[i - 300] + bhh2[i - 300];
    sW[i] = v;
  }
  __syncthreads();

  const int b = blockIdx.x * blockDim.x + threadIdx.x;  // grid covers exactly B
  const float* xb = x + (size_t)b * (T_ * IN_);
  f16* Fb = F + (size_t)b * KP_;

  float h0[3] = {0,0,0}, c0[3] = {0,0,0};
  float h1[3] = {0,0,0}, c1[3] = {0,0,0};
  float h2[3] = {0,0,0}, c2[3] = {0,0,0};

  for (int t = 0; t < T_; ++t) {
    const float4* xp = (const float4*)(xb + t * IN_);
    float4 xa = xp[0], xbv = xp[1];
    float xin[8] = {xa.x, xa.y, xa.z, xa.w, xbv.x, xbv.y, xbv.z, xbv.w};

    float g[12];
    // layer 0: din = 8
#pragma unroll
    for (int r = 0; r < 12; ++r) {
      float a = sW[132 + r];
#pragma unroll
      for (int k = 0; k < 8; ++k) a += xin[k] * sW[r*8 + k];
#pragma unroll
      for (int j = 0; j < 3; ++j) a += h0[j] * sW[96 + r*3 + j];
      g[r] = a;
    }
    lstm_cell(g, h0, c0);
    // layer 1: din = 3 (input h0)
#pragma unroll
    for (int r = 0; r < 12; ++r) {
      float a = sW[216 + r];
#pragma unroll
      for (int k = 0; k < 3; ++k) a += h0[k] * sW[144 + r*3 + k];
#pragma unroll
      for (int j = 0; j < 3; ++j) a += h1[j] * sW[180 + r*3 + j];
      g[r] = a;
    }
    lstm_cell(g, h1, c1);
    // layer 2: din = 3 (input h1)
#pragma unroll
    for (int r = 0; r < 12; ++r) {
      float a = sW[300 + r];
#pragma unroll
      for (int k = 0; k < 3; ++k) a += h1[k] * sW[228 + r*3 + k];
#pragma unroll
      for (int j = 0; j < 3; ++j) a += h2[j] * sW[264 + r*3 + j];
      g[r] = a;
    }
    lstm_cell(g, h2, c2);

    Fb[t*3 + 0] = (f16)h2[0];
    Fb[t*3 + 1] = (f16)h2[1];
    Fb[t*3 + 2] = (f16)h2[2];
  }
#pragma unroll
  for (int k = TH_; k < KP_; ++k) Fb[k] = (f16)0.f;   // zero K-pad for WMMA
}

// ---------------------------------------------------------------- WMMA MLP: 612 -> 50 -> 50 -> 612
// 4 waves / block, 1 batch tile (16 rows) per wave; Wh+Wo async-staged in LDS.
__global__ __launch_bounds__(128) void mlp_wmma(const f16* __restrict__ F,
    const f16* __restrict__ W1, const f16* __restrict__ Wh, const f16* __restrict__ Wo,
    const float* __restrict__ b1, const float* __restrict__ bh, const float* __restrict__ bo,
    float* __restrict__ out) {
  extern __shared__ char smem[];
  f16* sWh = (f16*)smem;                                 // 64 x 64
  f16* sWo = (f16*)(smem + SWH_BYTES);                   // 624 x 64
  f16* sY  = (f16*)(smem + SWH_BYTES + SWO_BYTES) + (threadIdx.x >> 5) * (16 * 64);

  const int tid  = threadIdx.x;
  const int lane = tid & 31;
  const int wave = tid >> 5;
  const int row  = lane & 15;
  const int hi   = (lane >> 4) & 1;
  const int bBase = (blockIdx.x * 4 + wave) * 16;

  // ---- kick off async staging of Wh (8KB) + Wo (78KB) into LDS; overlaps with fc1 below
  {
    const int WHV = (LINP_ * LINP_) / 8;  // 512  b128 vectors -> 4 per thread
    const int WOV = (NJP_ * LINP_) / 8;   // 4992 b128 vectors -> 39 per thread
#if ASYNC_LDS
    for (int i = tid; i < WHV; i += 128)
      __builtin_amdgcn_global_load_async_to_lds_b128(
          (gp128)(Wh + i * 8), (lp128)(sWh + i * 8), 0, 0);
    for (int i = tid; i < WOV; i += 128)
      __builtin_amdgcn_global_load_async_to_lds_b128(
          (gp128)(Wo + i * 8), (lp128)(sWo + i * 8), 0, 0);
#else
    for (int i = tid; i < WHV; i += 128) *(v8h*)(sWh + i * 8) = *(const v8h*)(Wh + i * 8);
    for (int i = tid; i < WOV; i += 128) *(v8h*)(sWo + i * 8) = *(const v8h*)(Wo + i * 8);
#endif
  }

  // 16-bit 16x32 A/B fragment: lanes 0-15 -> halves {K..K+7, K+16..K+23},
  // lanes 16-31 -> halves {K+8..K+15, K+24..K+31}; two aligned b128 loads.
  auto loadAB = [&](const f16* base, int ld, int r, int k0) -> v16h {
    const f16* p = base + (size_t)r * ld + k0 + hi * 8;
    v8h lo = *(const v8h*)p;
    v8h hh = *(const v8h*)(p + 16);
    v16h a;
#pragma unroll
    for (int i = 0; i < 8; ++i) { a[i] = lo[i]; a[8 + i] = hh[i]; }
    return a;
  };
  // WMMA D (f32 16x16) -> row-major f16 LDS tile, + bias
  auto storeY = [&](v8f acc, int nt, const float* bias, int blim) {
    int n = nt * 16 + row;
    float bv = (n < blim) ? bias[n] : 0.f;
#pragma unroll
    for (int r8 = 0; r8 < 8; ++r8) sY[(r8 + hi * 8) * 64 + n] = (f16)(acc[r8] + bv);
  };

  // ---- fc1: (16,640) x (640,64) -> Y1 (16,64)
  // Branchless ping-pong double buffer: even/odd fragment sets, no rotation copies.
  // The tail load of chunk 20 over-reads 64B past the logical row but stays inside
  // the workspace (F is followed by W1/Wh/Wo) and its value is never consumed.
  v8f a0 = {}, a1 = {}, a2 = {}, a3 = {};
  v16h aE  = loadAB(F,  KP_, bBase + row, 0);
  v16h b0E = loadAB(W1, KP_,      row, 0);
  v16h b1E = loadAB(W1, KP_, 16 + row, 0);
  v16h b2E = loadAB(W1, KP_, 32 + row, 0);
  v16h b3E = loadAB(W1, KP_, 48 + row, 0);
#pragma unroll 1
  for (int kc = 0; kc < 20; kc += 2) {
    const int k1 = (kc + 1) * 32;
    const int k2 = (kc + 2) * 32;
    v16h aO  = loadAB(F,  KP_, bBase + row, k1);
    v16h b0O = loadAB(W1, KP_,      row, k1);
    v16h b1O = loadAB(W1, KP_, 16 + row, k1);
    v16h b2O = loadAB(W1, KP_, 32 + row, k1);
    v16h b3O = loadAB(W1, KP_, 48 + row, k1);
    a0 = __builtin_amdgcn_wmma_f32_16x16x32_f16(false, aE, false, b0E, (short)0, a0, false, false);
    a1 = __builtin_amdgcn_wmma_f32_16x16x32_f16(false, aE, false, b1E, (short)0, a1, false, false);
    a2 = __builtin_amdgcn_wmma_f32_16x16x32_f16(false, aE, false, b2E, (short)0, a2, false, false);
    a3 = __builtin_amdgcn_wmma_f32_16x16x32_f16(false, aE, false, b3E, (short)0, a3, false, false);
    aE  = loadAB(F,  KP_, bBase + row, k2);
    b0E = loadAB(W1, KP_,      row, k2);
    b1E = loadAB(W1, KP_, 16 + row, k2);
    b2E = loadAB(W1, KP_, 32 + row, k2);
    b3E = loadAB(W1, KP_, 48 + row, k2);
    a0 = __builtin_amdgcn_wmma_f32_16x16x32_f16(false, aO, false, b0O, (short)0, a0, false, false);
    a1 = __builtin_amdgcn_wmma_f32_16x16x32_f16(false, aO, false, b1O, (short)0, a1, false, false);
    a2 = __builtin_amdgcn_wmma_f32_16x16x32_f16(false, aO, false, b2O, (short)0, a2, false, false);
    a3 = __builtin_amdgcn_wmma_f32_16x16x32_f16(false, aO, false, b3O, (short)0, a3, false, false);
  }

  // async staging must be complete (own wave) and visible (all waves) before LDS weight use
#if ASYNC_LDS
  __builtin_amdgcn_s_wait_asynccnt(0);
#endif
  __syncthreads();

  storeY(a0, 0, b1, LIN_); storeY(a1, 1, b1, LIN_);
  storeY(a2, 2, b1, LIN_); storeY(a3, 3, b1, LIN_);

  // ---- hid: (16,64) x (64,64) -> Y2 (16,64), B from LDS
  v16h y0 = loadAB(sY, 64, row, 0);
  v16h y1 = loadAB(sY, 64, row, 32);
  v8f h0a = {}, h1a = {}, h2a = {}, h3a = {};
  h0a = __builtin_amdgcn_wmma_f32_16x16x32_f16(false, y0, false, loadAB(sWh, 64,      row,  0), (short)0, h0a, false, false);
  h0a = __builtin_amdgcn_wmma_f32_16x16x32_f16(false, y1, false, loadAB(sWh, 64,      row, 32), (short)0, h0a, false, false);
  h1a = __builtin_amdgcn_wmma_f32_16x16x32_f16(false, y0, false, loadAB(sWh, 64, 16 + row,  0), (short)0, h1a, false, false);
  h1a = __builtin_amdgcn_wmma_f32_16x16x32_f16(false, y1, false, loadAB(sWh, 64, 16 + row, 32), (short)0, h1a, false, false);
  h2a = __builtin_amdgcn_wmma_f32_16x16x32_f16(false, y0, false, loadAB(sWh, 64, 32 + row,  0), (short)0, h2a, false, false);
  h2a = __builtin_amdgcn_wmma_f32_16x16x32_f16(false, y1, false, loadAB(sWh, 64, 32 + row, 32), (short)0, h2a, false, false);
  h3a = __builtin_amdgcn_wmma_f32_16x16x32_f16(false, y0, false, loadAB(sWh, 64, 48 + row,  0), (short)0, h3a, false, false);
  h3a = __builtin_amdgcn_wmma_f32_16x16x32_f16(false, y1, false, loadAB(sWh, 64, 48 + row, 32), (short)0, h3a, false, false);

  storeY(h0a, 0, bh, LIN_); storeY(h1a, 1, bh, LIN_);
  storeY(h2a, 2, bh, LIN_); storeY(h3a, 3, bh, LIN_);

  // ---- out: (16,64) x (64,624) -> (16,612) + bias, B from LDS, f32 store
  v16h z0 = loadAB(sY, 64, row, 0);
  v16h z1 = loadAB(sY, 64, row, 32);
#pragma unroll 1
  for (int jt = 0; jt < 39; ++jt) {
    v8f acc = {};
    acc = __builtin_amdgcn_wmma_f32_16x16x32_f16(false, z0, false, loadAB(sWo, 64, jt * 16 + row,  0), (short)0, acc, false, false);
    acc = __builtin_amdgcn_wmma_f32_16x16x32_f16(false, z1, false, loadAB(sWo, 64, jt * 16 + row, 32), (short)0, acc, false, false);
    int j = jt * 16 + row;
    if (j < TH_) {
      float bv = bo[j];
#pragma unroll
      for (int r8 = 0; r8 < 8; ++r8)
        out[(size_t)(bBase + r8 + hi * 8) * TH_ + j] = acc[r8] + bv;
    }
  }
}

// ---------------------------------------------------------------- launch
extern "C" void kernel_launch(void* const* d_in, const int* in_sizes, int n_in,
                              void* d_out, int out_size, void* d_ws, size_t ws_size,
                              hipStream_t stream) {
  const float* x    = (const float*)d_in[0];
  const float* Wih0 = (const float*)d_in[1];
  const float* Whh0 = (const float*)d_in[2];
  const float* bih0 = (const float*)d_in[3];
  const float* bhh0 = (const float*)d_in[4];
  const float* Wih1 = (const float*)d_in[5];
  const float* Whh1 = (const float*)d_in[6];
  const float* bih1 = (const float*)d_in[7];
  const float* bhh1 = (const float*)d_in[8];
  const float* Wih2 = (const float*)d_in[9];
  const float* Whh2 = (const float*)d_in[10];
  const float* bih2 = (const float*)d_in[11];
  const float* bhh2 = (const float*)d_in[12];
  const float* Wfc1 = (const float*)d_in[13];
  const float* bfc1 = (const float*)d_in[14];
  const float* Whid = (const float*)d_in[15];
  const float* bhid = (const float*)d_in[16];
  const float* Wout = (const float*)d_in[17];
  const float* bout = (const float*)d_in[18];
  float* out = (float*)d_out;

  char* ws = (char*)d_ws;
  size_t off = 0;
  f16* F  = (f16*)(ws + off); off += (size_t)B_ * KP_ * sizeof(f16);   // 20.97 MB
  f16* W1 = (f16*)(ws + off); off += (size_t)LINP_ * KP_ * sizeof(f16);
  f16* Wh = (f16*)(ws + off); off += (size_t)LINP_ * LINP_ * sizeof(f16);
  f16* Wo = (f16*)(ws + off); off += (size_t)NJP_ * LINP_ * sizeof(f16);

  prep_weights<<<128, 256, 0, stream>>>(Wfc1, Whid, Wout, W1, Wh, Wo);
  lstm3<<<B_ / 64, 64, 0, stream>>>(x, Wih0, Whh0, bih0, bhh0,
                                    Wih1, Whh1, bih1, bhh1,
                                    Wih2, Whh2, bih2, bhh2, F);
  mlp_wmma<<<B_ / 64, 128, SMEM_BYTES, stream>>>(F, W1, Wh, Wo, bfc1, bhid, bout, out);
}
